// BiLSTMWithLM_13735305413416
// MI455X (gfx1250) — matrix operand order repaired and checked by reading, not compiled, not run on hardware
//
#include <hip/hip_runtime.h>
#include <hip/hip_bf16.h>

// ---------------------------------------------------------------------------
// BiLSTM + LM head for MI455X (gfx1250, wave32, WMMA bf16)
//  * Batched input-to-hidden GEMMs: 32x64 wave tiles, branch-free software
//    pipeline with two statically-named fragment buffer sets.
//  * amdgpu_waves_per_eu(2) on the WMMA kernels (256 thr = 2 waves/SIMD)
//    raises the VGPR cap to 512/wave -> no scratch spills in inner loops.
//  * Recurrent h@W_hh^T: persistent kernel (1 WG/direction), h in LDS,
//    W_hh fragments register-resident, gates i/f/g/o co-located per lane.
//  * global_prefetch of next timestep's pre-gates overlaps DMA with compute.
// ---------------------------------------------------------------------------

typedef __attribute__((ext_vector_type(16))) __bf16 v16bf;
typedef __attribute__((ext_vector_type(8)))  float  v8f;

#define Tn    1024
#define Bn    64
#define DIN   400
#define KP0   416          // DIN padded to multiple of 32
#define Hn    128
#define H4    512          // 4*H (gates)
#define HD1   256
#define HD2   64
#define Mrows (Tn*Bn)      // 65536

// ---------------------------------------------------------------------------
// WMMA fragment loaders (lane layouts per CDNA5 ISA 7.12.2)
// ---------------------------------------------------------------------------

// A fragment 16x32 bf16 from row-major [*, lda].
// Lane L (r=L&15, h=L>>4) holds row r, k in [h*8,h*8+8) and [16+h*8,16+h*8+8).
__device__ __forceinline__ v16bf load_a_frag(const __bf16* __restrict__ base, int lda) {
    const int lane = threadIdx.x & 31;
    const int r    = lane & 15;
    const int half = lane >> 4;
    const __bf16* p = base + (size_t)r * lda + half * 8;
    v16bf a;
#pragma unroll
    for (int q = 0; q < 8; ++q) a[q] = p[q];
#pragma unroll
    for (int q = 0; q < 8; ++q) a[8 + q] = p[16 + q];
    return a;
}

// B fragment from pre-packed layout: 16 contiguous bf16 per lane.
__device__ __forceinline__ v16bf load_b_frag(const __bf16* __restrict__ p) {
    const int lane = threadIdx.x & 31;
    const __bf16* q = p + lane * 16;
    v16bf b;
#pragma unroll
    for (int i = 0; i < 16; ++i) b[i] = q[i];
    return b;
}

__device__ __forceinline__ float sigm(float x) { return 1.f / (1.f + __expf(-x)); }

// ---------------------------------------------------------------------------
// Weight packing: W [N][Ksrc] row-major -> B = W^T fragments
// packed[((nt*KT + kt)*32 + lane)*16 + q] = W[n][k]
//   n = nt*16 + (lane&15),  k = kt*32 + (q>>3)*16 + (lane>>4)*8 + (q&7)   (zero-padded)
// ---------------------------------------------------------------------------
__global__ void pack_b_kernel(const float* __restrict__ W, __bf16* __restrict__ out,
                              int N, int Ksrc, int Kpad) {
    const int KT = Kpad >> 5;
    const size_t total = (size_t)(N >> 4) * KT * 512;
    size_t i = (size_t)blockIdx.x * blockDim.x + threadIdx.x;
    if (i >= total) return;
    const int q    = (int)(i & 15);
    const int lane = (int)((i >> 4) & 31);
    const size_t r = i >> 9;
    const int kt = (int)(r % KT);
    const int nt = (int)(r / KT);
    const int n = nt * 16 + (lane & 15);
    const int k = kt * 32 + (q >> 3) * 16 + (lane >> 4) * 8 + (q & 7);
    float v = (n < N && k < Ksrc) ? W[(size_t)n * Ksrc + k] : 0.f;
    out[i] = (__bf16)v;
}

// x [B][T][DIN] fp32 -> xb [(t*B+b)][KP0] bf16, zero padded
__global__ void x_to_bf16_kernel(const float* __restrict__ x, __bf16* __restrict__ xb) {
    size_t i = (size_t)blockIdx.x * blockDim.x + threadIdx.x;
    const size_t total = (size_t)Mrows * KP0;
    if (i >= total) return;
    const int d = (int)(i % KP0);
    const size_t r = i / KP0;       // r = t*B + b
    const int t = (int)(r / Bn);
    const int b = (int)(r % Bn);
    __bf16 v = (__bf16)0.f;
    if (d < DIN) v = (__bf16)x[((size_t)b * Tn + t) * DIN + d];
    xb[i] = v;
}

// ---------------------------------------------------------------------------
// WMMA GEMM: C[M][N] f32 = A_bf16[M][K] @ Bpacked + bias0 + bias1
// One wave computes a 32x64 tile. Branch-free double-buffered pipeline;
// 1/2/3-ktile epilogue handled outside the steady-state loop.
// ---------------------------------------------------------------------------
#define GEMM_LOAD_A(dst0, dst1, ktv)                                            \
    do {                                                                        \
        dst0 = load_a_frag(A + (size_t)m0 * K + (ktv) * 32, K);                 \
        dst1 = load_a_frag(A + (size_t)(m0 + 16) * K + (ktv) * 32, K);          \
    } while (0)

#define GEMM_LOAD_B(dst, ktv)                                                   \
    do {                                                                        \
        _Pragma("unroll")                                                       \
        for (int j = 0; j < 4; ++j)                                             \
            dst[j] = load_b_frag(Bp + (size_t)((ng * 4 + j) * KT + (ktv)) * 512);\
    } while (0)

#define GEMM_MMA(aa0, aa1, bb)                                                  \
    do {                                                                        \
        _Pragma("unroll")                                                       \
        for (int j = 0; j < 4; ++j)                                             \
            acc0[j] = __builtin_amdgcn_wmma_f32_16x16x32_bf16(                  \
                false, aa0, false, bb[j], (short)0, acc0[j], false, false);     \
        _Pragma("unroll")                                                       \
        for (int j = 0; j < 4; ++j)                                             \
            acc1[j] = __builtin_amdgcn_wmma_f32_16x16x32_bf16(                  \
                false, aa1, false, bb[j], (short)0, acc1[j], false, false);     \
    } while (0)

__global__ void __launch_bounds__(256)
__attribute__((amdgpu_waves_per_eu(2)))
gemm_bf16_kernel(const __bf16* __restrict__ A,
                 const __bf16* __restrict__ Bp,
                 const float* __restrict__ bias0,
                 const float* __restrict__ bias1,
                 float* __restrict__ C,
                 int M, int K, int N) {
    const int lane = threadIdx.x & 31;
    const int wid  = blockIdx.x * (blockDim.x >> 5) + (threadIdx.x >> 5);
    const int ngrp = N >> 6;                       // #64-wide column groups
    const int wtot = (M >> 5) * ngrp;
    if (wid >= wtot) return;
    const int m0 = (wid / ngrp) << 5;              // 32 rows per wave
    const int ng = wid % ngrp;
    const int KT = K >> 5;

    v8f acc0[4] = {};    // rows [m0, m0+16)
    v8f acc1[4] = {};    // rows [m0+16, m0+32)
    v16bf a00, a01, b0[4];   // pipeline buffer 0
    v16bf a10, a11, b1[4];   // pipeline buffer 1

    GEMM_LOAD_A(a00, a01, 0);
    GEMM_LOAD_B(b0, 0);

    int kt = 0;
    // steady state: a next pair always exists -> no branches in body
    for (; kt + 3 < KT; kt += 2) {
        GEMM_LOAD_A(a10, a11, kt + 1);
        GEMM_LOAD_B(b1, kt + 1);
        GEMM_MMA(a00, a01, b0);
        GEMM_LOAD_A(a00, a01, kt + 2);
        GEMM_LOAD_B(b0, kt + 2);
        GEMM_MMA(a10, a11, b1);
    }
    // epilogue: 1, 2 or 3 remaining k-tiles (buf0 holds tile kt)
    const int rem = KT - kt;
    if (rem == 3) {
        GEMM_LOAD_A(a10, a11, kt + 1);
        GEMM_LOAD_B(b1, kt + 1);
        GEMM_MMA(a00, a01, b0);
        GEMM_LOAD_A(a00, a01, kt + 2);
        GEMM_LOAD_B(b0, kt + 2);
        GEMM_MMA(a10, a11, b1);
        GEMM_MMA(a00, a01, b0);
    } else if (rem == 2) {
        GEMM_LOAD_A(a10, a11, kt + 1);
        GEMM_LOAD_B(b1, kt + 1);
        GEMM_MMA(a00, a01, b0);
        GEMM_MMA(a10, a11, b1);
    } else {
        GEMM_MMA(a00, a01, b0);
    }

    const int half = lane >> 4;
    const int nn   = lane & 15;
#pragma unroll
    for (int j = 0; j < 4; ++j) {
        const int n0 = (ng * 4 + j) * 16 + nn;
        float bb = bias0 ? bias0[n0] : 0.f;
        if (bias1) bb += bias1[n0];
#pragma unroll
        for (int i = 0; i < 8; ++i) {
            C[(size_t)(m0 + half * 8 + i) * N + n0]      = acc0[j][i] + bb;
            C[(size_t)(m0 + 16 + half * 8 + i) * N + n0] = acc1[j][i] + bb;
        }
    }
}

// ---------------------------------------------------------------------------
// Persistent recurrent kernel. grid = 2 blocks (dir 0 fwd, dir 1 bwd),
// blockDim = 256 (8 waves = 2 waves/SIMD). Wave w owns output units
// j in [16w, 16w+16) for ALL four gates -> gate fusion in-register.
// ~300 VGPRs live (acc 128 + W_hh frags 128 + state) -> waves_per_eu(2).
// ---------------------------------------------------------------------------
__global__ void __launch_bounds__(256)
__attribute__((amdgpu_waves_per_eu(2)))
lstm_recur_kernel(const float* __restrict__ pre_f,
                  const float* __restrict__ pre_b,
                  const __bf16* __restrict__ whh_f,
                  const __bf16* __restrict__ whh_b,
                  __bf16* __restrict__ hcat) {
    const int dir = blockIdx.x;
    const float*  pre  = dir ? pre_b : pre_f;
    const __bf16* whhp = dir ? whh_b : whh_f;

    __shared__ __bf16 h_lds[Bn * Hn];   // 16 KB

    const int tid  = threadIdx.x;
    const int lane = tid & 31;
    const int w    = tid >> 5;          // 0..7
    const int half = lane >> 4;
    const int nn   = lane & 15;

    for (int i = tid; i < Bn * Hn; i += blockDim.x) h_lds[i] = (__bf16)0.f;

    // Preload W_hh^T fragments: gate g tile index = g*8 + w, K tiles 0..3.
    v16bf bfr[4][4];
#pragma unroll
    for (int g = 0; g < 4; ++g)
#pragma unroll
        for (int kt = 0; kt < 4; ++kt)
            bfr[g][kt] = load_b_frag(whhp + ((size_t)((g * 8 + w) * 4 + kt)) * 512);

    v8f cst[4] = {};                    // cell state c for (m, j) per lane slot
    __syncthreads();

    for (int s = 0; s < Tn; ++s) {
        const int t = dir ? (Tn - 1 - s) : s;
        const float* pt = pre + (size_t)t * Bn * H4;

        // init accumulators from precomputed x@W_ih + biases
        v8f acc[4][4];
#pragma unroll
        for (int mt = 0; mt < 4; ++mt)
#pragma unroll
            for (int g = 0; g < 4; ++g) {
                const float* p = pt + (size_t)(mt * 16 + half * 8) * H4 + g * Hn + w * 16 + nn;
#pragma unroll
                for (int i = 0; i < 8; ++i) acc[mt][g][i] = p[(size_t)i * H4];
            }

        // prefetch next timestep's pre-gate tile into caches (global_prefetch)
        if (s + 1 < Tn) {
            const int t2 = dir ? (Tn - 2 - s) : (s + 1);
            const float* pn = pre + (size_t)t2 * Bn * H4;
#pragma unroll
            for (int mt = 0; mt < 4; ++mt)
                __builtin_prefetch(pn + (size_t)(mt * 16 + half * 8) * H4 + w * 16 + nn, 0, 1);
        }

        // gates += h @ W_hh^T  (64 WMMAs per wave per step)
#pragma unroll
        for (int mt = 0; mt < 4; ++mt)
#pragma unroll
            for (int kt = 0; kt < 4; ++kt) {
                v16bf a = load_a_frag(h_lds + mt * 16 * Hn + kt * 32, Hn);
#pragma unroll
                for (int g = 0; g < 4; ++g)
                    acc[mt][g] = __builtin_amdgcn_wmma_f32_16x16x32_bf16(
                        false, a, false, bfr[g][kt], (short)0, acc[mt][g], false, false);
            }

        __syncthreads();   // all waves done reading h_lds for this step

        __bf16* outp = hcat + (size_t)t * Bn * (2 * Hn) + dir * Hn;
#pragma unroll
        for (int mt = 0; mt < 4; ++mt) {
#pragma unroll
            for (int i = 0; i < 8; ++i) {
                const float ig = acc[mt][0][i];
                const float fg = acc[mt][1][i];
                const float gg = acc[mt][2][i];
                const float og = acc[mt][3][i];
                float c = sigm(fg) * cst[mt][i] + sigm(ig) * tanhf(gg);
                float h = sigm(og) * tanhf(c);
                cst[mt][i] = c;
                const int m = mt * 16 + half * 8 + i;
                const int j = w * 16 + nn;
                const __bf16 hb = (__bf16)h;
                h_lds[m * Hn + j] = hb;
                outp[(size_t)m * (2 * Hn) + j] = hb;
            }
        }
        __syncthreads();
    }
}

// BN1 fused into bf16 conversion: abn = bn1(hcat1) as bf16 for Linear GEMM
__global__ void bn1_to_bf16_kernel(const __bf16* __restrict__ in, __bf16* __restrict__ out,
                                   const float* __restrict__ g, const float* __restrict__ b,
                                   const float* __restrict__ m, const float* __restrict__ v) {
    size_t i = (size_t)blockIdx.x * blockDim.x + threadIdx.x;
    const size_t total = (size_t)Mrows * HD1;
    if (i >= total) return;
    const int c = (int)(i & (HD1 - 1));
    const float x = (float)in[i];
    out[i] = (__bf16)((x - m[c]) * rsqrtf(v[c] + 1e-5f) * g[c] + b[c]);
}

// tanh then BN2, in place on h2 [Mrows][HD2] f32
__global__ void tanh_bn2_kernel(float* __restrict__ h2,
                                const float* __restrict__ g, const float* __restrict__ b,
                                const float* __restrict__ m, const float* __restrict__ v) {
    size_t i = (size_t)blockIdx.x * blockDim.x + threadIdx.x;
    const size_t total = (size_t)Mrows * HD2;
    if (i >= total) return;
    const int c = (int)(i & (HD2 - 1));
    const float x = tanhf(h2[i]);
    h2[i] = (x - m[c]) * rsqrtf(v[c] + 1e-5f) * g[c] + b[c];
}

// Autoregressive 2-class head: 1 block, 64 threads, thread = batch row.
__global__ void ar_head_kernel(const float* __restrict__ h2,
                               const float* __restrict__ ow,   // [2][68]
                               const float* __restrict__ ob,   // [2]
                               float* __restrict__ out) {      // [B*T][2]
    __shared__ float w[2 * 68 + 2];
    const int tid = threadIdx.x;
    for (int i = tid; i < 136; i += blockDim.x) w[i] = ow[i];
    if (tid < 2) w[136 + tid] = ob[tid];
    __syncthreads();
    if (tid >= Bn) return;
    const int b = tid;
    float c0 = 0.f, c1 = 0.f, c2 = 0.f, c3 = 0.f;
    for (int t = 0; t < Tn; ++t) {
        const float* h = h2 + ((size_t)t * Bn + b) * HD2;
        float l0 = w[136] + c0 * w[0]  + c1 * w[1]  + c2 * w[2]  + c3 * w[3];
        float l1 = w[137] + c0 * w[68] + c1 * w[69] + c2 * w[70] + c3 * w[71];
#pragma unroll 8
        for (int j = 0; j < HD2; ++j) {
            const float hv = h[j];
            l0 += hv * w[4 + j];
            l1 += hv * w[72 + j];
        }
        const float mx  = fmaxf(l0, l1);
        const float lse = mx + __logf(__expf(l0 - mx) + __expf(l1 - mx));
        const float lo0 = l0 - lse, lo1 = l1 - lse;
        out[((size_t)b * Tn + t) * 2 + 0] = lo0;
        out[((size_t)b * Tn + t) * 2 + 1] = lo1;
        c0 = c2; c1 = c3; c2 = lo0; c3 = lo1;
    }
}

// ---------------------------------------------------------------------------
// Host launcher
// ---------------------------------------------------------------------------
static inline size_t bump(size_t& o, size_t bytes) {
    size_t r = o;
    o = (o + bytes + 255) & ~(size_t)255;
    return r;
}

extern "C" void kernel_launch(void* const* d_in, const int* in_sizes, int n_in,
                              void* d_out, int out_size, void* d_ws, size_t ws_size,
                              hipStream_t stream) {
    (void)in_sizes; (void)n_in; (void)out_size; (void)ws_size;
    const float* x        = (const float*)d_in[0];
    const float* w_ih_l0  = (const float*)d_in[1];
    const float* w_hh_l0  = (const float*)d_in[2];
    const float* b_ih_l0  = (const float*)d_in[3];
    const float* b_hh_l0  = (const float*)d_in[4];
    const float* w_ih_l0r = (const float*)d_in[5];
    const float* w_hh_l0r = (const float*)d_in[6];
    const float* b_ih_l0r = (const float*)d_in[7];
    const float* b_hh_l0r = (const float*)d_in[8];
    const float* w_ih_l1  = (const float*)d_in[9];
    const float* w_hh_l1  = (const float*)d_in[10];
    const float* b_ih_l1  = (const float*)d_in[11];
    const float* b_hh_l1  = (const float*)d_in[12];
    const float* w_ih_l1r = (const float*)d_in[13];
    const float* w_hh_l1r = (const float*)d_in[14];
    const float* b_ih_l1r = (const float*)d_in[15];
    const float* b_hh_l1r = (const float*)d_in[16];
    const float* bn1_g = (const float*)d_in[17];
    const float* bn1_b = (const float*)d_in[18];
    const float* bn1_m = (const float*)d_in[19];
    const float* bn1_v = (const float*)d_in[20];
    const float* lin_w = (const float*)d_in[21];
    const float* lin_b = (const float*)d_in[22];
    const float* bn2_g = (const float*)d_in[23];
    const float* bn2_b = (const float*)d_in[24];
    const float* bn2_m = (const float*)d_in[25];
    const float* bn2_v = (const float*)d_in[26];
    const float* out_w = (const float*)d_in[27];
    const float* out_b = (const float*)d_in[28];

    char* ws = (char*)d_ws;
    size_t o = 0;
    // activations
    __bf16* xb     = (__bf16*)(ws + bump(o, (size_t)Mrows * KP0 * 2));
    float*  pre_f  = (float*) (ws + bump(o, (size_t)Mrows * H4  * 4));
    float*  pre_b  = (float*) (ws + bump(o, (size_t)Mrows * H4  * 4));
    __bf16* hcat0  = (__bf16*)(ws + bump(o, (size_t)Mrows * HD1 * 2));
    __bf16* hcat1  = (__bf16*)(ws + bump(o, (size_t)Mrows * HD1 * 2));
    __bf16* abn    = (__bf16*)(ws + bump(o, (size_t)Mrows * HD1 * 2));
    float*  h2     = (float*) (ws + bump(o, (size_t)Mrows * HD2 * 4));
    // packed weights (B-fragment layout)
    const size_t szp_ih0 = (size_t)(H4 / 16) * (KP0 / 32) * 512;  // elems
    const size_t szp_ih1 = (size_t)(H4 / 16) * (HD1 / 32) * 512;
    const size_t szp_hh  = (size_t)(H4 / 16) * (Hn  / 32) * 512;
    const size_t szp_lin = (size_t)(HD2 / 16) * (HD1 / 32) * 512;
    __bf16* wp_ih0f = (__bf16*)(ws + bump(o, szp_ih0 * 2));
    __bf16* wp_ih0b = (__bf16*)(ws + bump(o, szp_ih0 * 2));
    __bf16* wp_ih1f = (__bf16*)(ws + bump(o, szp_ih1 * 2));
    __bf16* wp_ih1b = (__bf16*)(ws + bump(o, szp_ih1 * 2));
    __bf16* wp_hh0f = (__bf16*)(ws + bump(o, szp_hh * 2));
    __bf16* wp_hh0b = (__bf16*)(ws + bump(o, szp_hh * 2));
    __bf16* wp_hh1f = (__bf16*)(ws + bump(o, szp_hh * 2));
    __bf16* wp_hh1b = (__bf16*)(ws + bump(o, szp_hh * 2));
    __bf16* wp_lin  = (__bf16*)(ws + bump(o, szp_lin * 2));

    const int TB = 256;
    auto blocks = [](size_t n, int tb) { return (unsigned)((n + tb - 1) / tb); };

    // --- weight packing ---
    pack_b_kernel<<<blocks(szp_ih0, TB), TB, 0, stream>>>(w_ih_l0,  wp_ih0f, H4, DIN, KP0);
    pack_b_kernel<<<blocks(szp_ih0, TB), TB, 0, stream>>>(w_ih_l0r, wp_ih0b, H4, DIN, KP0);
    pack_b_kernel<<<blocks(szp_ih1, TB), TB, 0, stream>>>(w_ih_l1,  wp_ih1f, H4, HD1, HD1);
    pack_b_kernel<<<blocks(szp_ih1, TB), TB, 0, stream>>>(w_ih_l1r, wp_ih1b, H4, HD1, HD1);
    pack_b_kernel<<<blocks(szp_hh,  TB), TB, 0, stream>>>(w_hh_l0,  wp_hh0f, H4, Hn, Hn);
    pack_b_kernel<<<blocks(szp_hh,  TB), TB, 0, stream>>>(w_hh_l0r, wp_hh0b, H4, Hn, Hn);
    pack_b_kernel<<<blocks(szp_hh,  TB), TB, 0, stream>>>(w_hh_l1,  wp_hh1f, H4, Hn, Hn);
    pack_b_kernel<<<blocks(szp_hh,  TB), TB, 0, stream>>>(w_hh_l1r, wp_hh1b, H4, Hn, Hn);
    pack_b_kernel<<<blocks(szp_lin, TB), TB, 0, stream>>>(lin_w,    wp_lin,  HD2, HD1, HD1);

    // --- x -> bf16 (time-major, K-padded) ---
    x_to_bf16_kernel<<<blocks((size_t)Mrows * KP0, TB), TB, 0, stream>>>(x, xb);

    // --- layer 0 pre-gates: [65536,416] x [416,512] ---
    {
        const unsigned nb = (unsigned)(((size_t)(Mrows / 32) * (H4 / 64)) / 8);
        gemm_bf16_kernel<<<nb, TB, 0, stream>>>(xb, wp_ih0f, b_ih_l0,  b_hh_l0,  pre_f, Mrows, KP0, H4);
        gemm_bf16_kernel<<<nb, TB, 0, stream>>>(xb, wp_ih0b, b_ih_l0r, b_hh_l0r, pre_b, Mrows, KP0, H4);
    }
    // --- layer 0 recurrence (both directions in parallel) ---
    lstm_recur_kernel<<<2, 256, 0, stream>>>(pre_f, pre_b, wp_hh0f, wp_hh0b, hcat0);

    // --- layer 1 pre-gates: [65536,256] x [256,512] (reuse pre buffers) ---
    {
        const unsigned nb = (unsigned)(((size_t)(Mrows / 32) * (H4 / 64)) / 8);
        gemm_bf16_kernel<<<nb, TB, 0, stream>>>(hcat0, wp_ih1f, b_ih_l1,  b_hh_l1,  pre_f, Mrows, HD1, H4);
        gemm_bf16_kernel<<<nb, TB, 0, stream>>>(hcat0, wp_ih1b, b_ih_l1r, b_hh_l1r, pre_b, Mrows, HD1, H4);
    }
    // --- layer 1 recurrence ---
    lstm_recur_kernel<<<2, 256, 0, stream>>>(pre_f, pre_b, wp_hh1f, wp_hh1b, hcat1);

    // --- BN1 -> bf16 ---
    bn1_to_bf16_kernel<<<blocks((size_t)Mrows * HD1, TB), TB, 0, stream>>>(
        hcat1, abn, bn1_g, bn1_b, bn1_m, bn1_v);

    // --- Linear [65536,256] x [256,64] ---
    {
        const unsigned nb = (unsigned)(((size_t)(Mrows / 32) * (HD2 / 64)) / 8);
        gemm_bf16_kernel<<<nb, TB, 0, stream>>>(abn, wp_lin, lin_b, nullptr, h2, Mrows, HD1, HD2);
    }
    // --- tanh + BN2 ---
    tanh_bn2_kernel<<<blocks((size_t)Mrows * HD2, TB), TB, 0, stream>>>(
        h2, bn2_g, bn2_b, bn2_m, bn2_v);

    // --- autoregressive log-softmax head ---
    ar_head_kernel<<<1, 64, 0, stream>>>(h2, out_w, out_b, (float*)d_out);
}
